// MultiHeadAttentionModule_73864847556658
// MI455X (gfx1250) — matrix operand order repaired and checked by reading, not compiled, not run on hardware
//
#include <hip/hip_runtime.h>
#include <hip/hip_bf16.h>
#include <math.h>

// ---- problem constants ----
#define SEQ     512
#define DMODEL  512
#define NHEAD   8
#define DHEAD   64
#define BT      32                 // B*T
#define MROWS   (BT * SEQ)         // 16384

typedef _Float16 h16;
typedef __attribute__((ext_vector_type(8)))  _Float16 h16x8;
typedef __attribute__((ext_vector_type(16))) _Float16 h16x16;
typedef __attribute__((ext_vector_type(8)))  float    f32x8;

// ---------------------------------------------------------------------------
// WMMA helpers (CDNA5 wave32, v_wmma_f32_16x16x32_f16)
// ---------------------------------------------------------------------------
__device__ __forceinline__ f32x8 wmma16(h16x16 a, h16x16 b, f32x8 c) {
  // (neg_a, A, neg_b, B, c_mod, C, reuse_a, reuse_b)
  return __builtin_amdgcn_wmma_f32_16x16x32_f16(false, a, false, b, (short)0, c, false, false);
}

// A-fragment: 16x32 f16 tile, row-major source (row M, 32 contiguous K).
// Lane l: M = l&15; holds K = c0..c0+7 and c0+16..c0+23 with c0 = (l>>4)*8.
__device__ __forceinline__ h16x16 frag_a(const h16* p, long ld, int lane) {
  const h16* q = p + (long)(lane & 15) * ld + ((lane >> 4) << 3);
  h16x8 lo = *(const h16x8*)q;
  h16x8 hi = *(const h16x8*)(q + 16);
  return __builtin_shufflevector(lo, hi, 0,1,2,3,4,5,6,7,8,9,10,11,12,13,14,15);
}

// B-fragment: 32x16 f16 tile B[k][n], sourced from memory where row index == n
// and K is contiguous (i.e. a transposed/row-per-column layout).
// Lane l: n = l&15; holds K = c0..c0+7 and c0+16..c0+23 with c0 = (l>>4)*8.
__device__ __forceinline__ h16x16 frag_b(const h16* p, long ld, int lane) {
  const h16* q = p + (long)(lane & 15) * ld + ((lane >> 4) << 3);
  h16x8 lo = *(const h16x8*)q;
  h16x8 hi = *(const h16x8*)(q + 16);
  return __builtin_shufflevector(lo, hi, 0,1,2,3,4,5,6,7,8,9,10,11,12,13,14,15);
}

// ---------------------------------------------------------------------------
// LayerNorm (eps=1e-3) + cast to f16.  One row (512) per 256-thread block.
// ---------------------------------------------------------------------------
__global__ void __launch_bounds__(256) layernorm_kernel(
    const float* __restrict__ in, const float* __restrict__ g,
    const float* __restrict__ b, h16* __restrict__ out) {
  __shared__ float red[256];
  const int row = blockIdx.x, t = threadIdx.x;
  const float* p = in + (long)row * DMODEL;
  float a0 = p[t], a1 = p[t + 256];
  red[t] = a0 + a1;
  __syncthreads();
  for (int s = 128; s > 0; s >>= 1) { if (t < s) red[t] += red[t + s]; __syncthreads(); }
  float mu = red[0] * (1.0f / DMODEL);
  __syncthreads();
  float d0 = a0 - mu, d1 = a1 - mu;
  red[t] = d0 * d0 + d1 * d1;
  __syncthreads();
  for (int s = 128; s > 0; s >>= 1) { if (t < s) red[t] += red[t + s]; __syncthreads(); }
  float rs = rsqrtf(red[0] * (1.0f / DMODEL) + 1e-3f);
  h16* o = out + (long)row * DMODEL;
  o[t]       = (h16)(d0 * rs * g[t] + b[t]);
  o[t + 256] = (h16)(d1 * rs * g[t + 256] + b[t + 256]);
}

// ---------------------------------------------------------------------------
// Sinusoidal positional encoding (S x DM), f16 output.
// ---------------------------------------------------------------------------
__global__ void __launch_bounds__(256) posenc_kernel(h16* __restrict__ pos) {
  int idx = blockIdx.x * 256 + threadIdx.x;
  int s = idx >> 9, d = idx & (DMODEL - 1);
  float e = (float)(2 * (d >> 1)) * (1.0f / DMODEL);
  float angle = (float)s * powf(10000.0f, -e);
  float v = (d & 1) ? cosf(angle) : sinf(angle);
  pos[idx] = (h16)v;
}

// ---------------------------------------------------------------------------
// Transpose + cast f32 -> f16 (weights: out[n][k] = in[k][n])
// ---------------------------------------------------------------------------
__global__ void __launch_bounds__(256) transpose_cast_f32(
    const float* __restrict__ in, h16* __restrict__ out, int R, int C) {
  __shared__ float tile[32][33];
  const int tx = threadIdx.x, ty = threadIdx.y;          // 32 x 8
  const int r0 = blockIdx.x * 32, c0 = blockIdx.y * 32;
  #pragma unroll
  for (int i = 0; i < 32; i += 8)
    tile[ty + i][tx] = in[(long)(r0 + ty + i) * C + c0 + tx];
  __syncthreads();
  #pragma unroll
  for (int i = 0; i < 32; i += 8)
    out[(long)(c0 + ty + i) * R + r0 + tx] = (h16)tile[tx][ty + i];
}

// Transpose f16 -> f16 (for V -> V^T)
__global__ void __launch_bounds__(256) transpose_h16(
    const h16* __restrict__ in, h16* __restrict__ out, int R, int C) {
  __shared__ h16 tile[32][33];
  const int tx = threadIdx.x, ty = threadIdx.y;
  const int r0 = blockIdx.x * 32, c0 = blockIdx.y * 32;
  #pragma unroll
  for (int i = 0; i < 32; i += 8)
    tile[ty + i][tx] = in[(long)(r0 + ty + i) * C + c0 + tx];
  __syncthreads();
  #pragma unroll
  for (int i = 0; i < 32; i += 8)
    out[(long)(c0 + ty + i) * R + r0 + tx] = tile[tx][ty + i];
}

// ---------------------------------------------------------------------------
// GEMM: C[M,N] = A[M,K] @ Wt[N,K]^T + bias + extra[m mod SEQ]
// 256 threads = 8 waves (4x2); wave computes 32x32 via 4 WMMA per k-step.
// Operand reuse comes from the 192MB L2 (whole working set resident).
// ---------------------------------------------------------------------------
__global__ void __launch_bounds__(256) gemm_h16_kernel(
    const h16* __restrict__ A, const h16* __restrict__ Wt,
    const float* __restrict__ bias, const h16* __restrict__ extra,
    h16* __restrict__ outh, float* __restrict__ outf,
    int M, int N, int K) {
  const int lane = threadIdx.x & 31;
  const int wid  = threadIdx.x >> 5;
  const int m0 = blockIdx.x * 128 + (wid & 3) * 32;
  const int n0 = blockIdx.y * 64  + (wid >> 2) * 32;
  f32x8 acc[2][2] = {};
  const h16* Ar0 = A  + (long)m0 * K;
  const h16* Ar1 = A  + (long)(m0 + 16) * K;
  const h16* Bc0 = Wt + (long)n0 * K;
  const h16* Bc1 = Wt + (long)(n0 + 16) * K;
  for (int k0 = 0; k0 < K; k0 += 32) {
    h16x16 a0 = frag_a(Ar0 + k0, K, lane);
    h16x16 a1 = frag_a(Ar1 + k0, K, lane);
    h16x16 b0 = frag_b(Bc0 + k0, K, lane);
    h16x16 b1 = frag_b(Bc1 + k0, K, lane);
    acc[0][0] = wmma16(a0, b0, acc[0][0]);
    acc[0][1] = wmma16(a0, b1, acc[0][1]);
    acc[1][0] = wmma16(a1, b0, acc[1][0]);
    acc[1][1] = wmma16(a1, b1, acc[1][1]);
  }
  const int half = lane >> 4, nn = lane & 15;
  #pragma unroll
  for (int i = 0; i < 2; ++i)
    #pragma unroll
    for (int j = 0; j < 2; ++j) {
      const int colg = n0 + j * 16 + nn;
      const float bv = bias ? bias[colg] : 0.0f;
      #pragma unroll
      for (int r = 0; r < 8; ++r) {
        const int rowg = m0 + i * 16 + r + half * 8;
        float v = acc[i][j][r] + bv;
        if (extra) v += (float)extra[(long)(rowg & (SEQ - 1)) * N + colg];
        const long idx = (long)rowg * N + colg;
        if (outh) outh[idx] = (h16)v;
        if (outf) outf[idx] = v;
      }
    }
}

// ---------------------------------------------------------------------------
// Flash attention per (b,t,h).  KP already holds K + positional bias, so
// scores = Q·KPᵀ / 8.  Block of 128 threads = 4 waves, each wave owns 16
// query rows and streams 32-key blocks with online softmax.
// ---------------------------------------------------------------------------
__global__ void __launch_bounds__(128) attention_kernel(
    const h16* __restrict__ Q, const h16* __restrict__ KP,
    const h16* __restrict__ Vt, h16* __restrict__ CTX) {
  __shared__ __align__(16) h16 plds[4][16 * 32];   // per-wave P tile
  const int lane = threadIdx.x & 31, w = threadIdx.x >> 5;
  const int half = lane >> 4, nn = lane & 15;
  const int bth = blockIdx.y;
  const int bt = bth >> 3, h = bth & 7;
  const int q0 = blockIdx.x * 64 + w * 16;          // query tile base in S
  const long rowbase = (long)bt * SEQ;
  const h16* Qb = Q  + (rowbase + q0) * DMODEL + h * DHEAD;
  const h16* Kb = KP + rowbase * DMODEL + h * DHEAD;
  const h16* Vb = Vt + (long)(h * DHEAD) * MROWS + rowbase;

  const h16x16 qa0 = frag_a(Qb, DMODEL, lane);        // dh 0..31
  const h16x16 qa1 = frag_a(Qb + 32, DMODEL, lane);   // dh 32..63

  f32x8 O0 = {}, O1 = {}, O2 = {}, O3 = {};
  float rmax[8], rsum[8];
  #pragma unroll
  for (int r = 0; r < 8; ++r) { rmax[r] = -1e30f; rsum[r] = 0.0f; }

  for (int k0 = 0; k0 < SEQ; k0 += 32) {
    if (k0 + 32 < SEQ)  // warm L2/L0 for the next key block
      __builtin_prefetch(Kb + (long)(k0 + 32 + nn) * DMODEL, 0, 0);

    // scores: two 16x16 tiles over this 32-key block, K-reduction dh=64
    f32x8 s0 = {}, s1 = {};
    s0 = wmma16(qa0, frag_b(Kb + (long)k0 * DMODEL,            DMODEL, lane), s0);
    s0 = wmma16(qa1, frag_b(Kb + (long)k0 * DMODEL + 32,       DMODEL, lane), s0);
    s1 = wmma16(qa0, frag_b(Kb + (long)(k0 + 16) * DMODEL,     DMODEL, lane), s1);
    s1 = wmma16(qa1, frag_b(Kb + (long)(k0 + 16) * DMODEL + 32,DMODEL, lane), s1);

    // online softmax per query row (row r+8*half, reduce over 16 lanes/half)
    #pragma unroll
    for (int r = 0; r < 8; ++r) {
      float x0 = s0[r] * 0.125f, x1 = s1[r] * 0.125f;
      float m = fmaxf(x0, x1);
      m = fmaxf(m, __shfl_xor(m, 1, 32));
      m = fmaxf(m, __shfl_xor(m, 2, 32));
      m = fmaxf(m, __shfl_xor(m, 4, 32));
      m = fmaxf(m, __shfl_xor(m, 8, 32));
      const float mnew = fmaxf(rmax[r], m);
      const float corr = __expf(rmax[r] - mnew);
      rmax[r] = mnew;
      const float p0 = __expf(x0 - mnew), p1 = __expf(x1 - mnew);
      rsum[r] = rsum[r] * corr + p0 + p1;   // lane-partial, reduced at end
      O0[r] *= corr; O1[r] *= corr; O2[r] *= corr; O3[r] *= corr;
      const int row = r + half * 8;
      plds[w][row * 32 + nn]      = (h16)p0;
      plds[w][row * 32 + 16 + nn] = (h16)p1;
    }
    asm volatile("s_wait_dscnt 0" ::: "memory");  // cross-lane LDS visibility

    // ctx += P(16x32) @ V(32keys x 64dh); V^T gives contiguous keys per dh row
    const h16x16 pa = frag_a(&plds[w][0], 32, lane);
    O0 = wmma16(pa, frag_b(Vb + 0L  * MROWS + k0, MROWS, lane), O0);
    O1 = wmma16(pa, frag_b(Vb + 16L * MROWS + k0, MROWS, lane), O1);
    O2 = wmma16(pa, frag_b(Vb + 32L * MROWS + k0, MROWS, lane), O2);
    O3 = wmma16(pa, frag_b(Vb + 48L * MROWS + k0, MROWS, lane), O3);
  }

  // normalize + store ctx as (b,t,s, h*64+dh) f16 for the output GEMM
  #pragma unroll
  for (int r = 0; r < 8; ++r) {
    float ssum = rsum[r];
    ssum += __shfl_xor(ssum, 1, 32);
    ssum += __shfl_xor(ssum, 2, 32);
    ssum += __shfl_xor(ssum, 4, 32);
    ssum += __shfl_xor(ssum, 8, 32);
    const float inv = 1.0f / ssum;
    const long rowg = rowbase + q0 + r + half * 8;
    h16* cp = CTX + rowg * DMODEL + h * DHEAD + nn;
    cp[0]  = (h16)(O0[r] * inv);
    cp[16] = (h16)(O1[r] * inv);
    cp[32] = (h16)(O2[r] * inv);
    cp[48] = (h16)(O3[r] * inv);
  }
}

// ---------------------------------------------------------------------------
// host launch
// ---------------------------------------------------------------------------
extern "C" void kernel_launch(void* const* d_in, const int* in_sizes, int n_in,
                              void* d_out, int out_size, void* d_ws, size_t ws_size,
                              hipStream_t stream) {
  (void)in_sizes; (void)n_in; (void)out_size; (void)ws_size;
  const float* x_in = (const float*)d_in[0];
  // d_in[1..4] (v,k,q,mask) are unused by the reference
  const float* ln_g = (const float*)d_in[5];
  const float* ln_b = (const float*)d_in[6];
  const float* wq_w = (const float*)d_in[7];
  const float* wq_b = (const float*)d_in[8];
  const float* wk_w = (const float*)d_in[9];
  const float* wk_b = (const float*)d_in[10];
  const float* wv_w = (const float*)d_in[11];
  const float* wv_b = (const float*)d_in[12];
  const float* wd_w = (const float*)d_in[13];
  const float* wd_b = (const float*)d_in[14];
  float* out = (float*)d_out;

  char* ws = (char*)d_ws;
  size_t off = 0;
  auto carve = [&](size_t elems) {
    h16* p = (h16*)(ws + off);
    off += ((elems * sizeof(h16)) + 255) & ~(size_t)255;
    return p;
  };
  h16* X    = carve((size_t)MROWS * DMODEL);   // layernormed input (f16)
  h16* Qh   = carve((size_t)MROWS * DMODEL);
  h16* KPh  = carve((size_t)MROWS * DMODEL);   // K + positional bias
  h16* Vh   = carve((size_t)MROWS * DMODEL);
  h16* VT   = carve((size_t)MROWS * DMODEL);   // V transposed
  h16* CTX  = carve((size_t)MROWS * DMODEL);
  h16* WqT  = carve((size_t)DMODEL * DMODEL);
  h16* WkT  = carve((size_t)DMODEL * DMODEL);
  h16* WvT  = carve((size_t)DMODEL * DMODEL);
  h16* WdT  = carve((size_t)DMODEL * DMODEL);
  h16* Pos  = carve((size_t)SEQ * DMODEL);
  h16* PosP = carve((size_t)SEQ * DMODEL);     // pos @ wd + wd_b

  const dim3 tb(32, 8);
  transpose_cast_f32<<<dim3(16, 16), tb, 0, stream>>>(wq_w, WqT, DMODEL, DMODEL);
  transpose_cast_f32<<<dim3(16, 16), tb, 0, stream>>>(wk_w, WkT, DMODEL, DMODEL);
  transpose_cast_f32<<<dim3(16, 16), tb, 0, stream>>>(wv_w, WvT, DMODEL, DMODEL);
  transpose_cast_f32<<<dim3(16, 16), tb, 0, stream>>>(wd_w, WdT, DMODEL, DMODEL);

  layernorm_kernel<<<MROWS, 256, 0, stream>>>(x_in, ln_g, ln_b, X);
  posenc_kernel<<<(SEQ * DMODEL) / 256, 256, 0, stream>>>(Pos);

  // pos projection (needed as 'extra' for the K GEMM)
  gemm_h16_kernel<<<dim3(SEQ / 128, DMODEL / 64), 256, 0, stream>>>(
      Pos, WdT, wd_b, nullptr, PosP, nullptr, SEQ, DMODEL, DMODEL);

  gemm_h16_kernel<<<dim3(MROWS / 128, DMODEL / 64), 256, 0, stream>>>(
      X, WqT, wq_b, nullptr, Qh, nullptr, MROWS, DMODEL, DMODEL);
  gemm_h16_kernel<<<dim3(MROWS / 128, DMODEL / 64), 256, 0, stream>>>(
      X, WkT, wk_b, PosP, KPh, nullptr, MROWS, DMODEL, DMODEL);  // K + pos bias
  gemm_h16_kernel<<<dim3(MROWS / 128, DMODEL / 64), 256, 0, stream>>>(
      X, WvT, wv_b, nullptr, Vh, nullptr, MROWS, DMODEL, DMODEL);

  transpose_h16<<<dim3(MROWS / 32, DMODEL / 32), tb, 0, stream>>>(Vh, VT, MROWS, DMODEL);

  attention_kernel<<<dim3(SEQ / 64, BT * NHEAD), 128, 0, stream>>>(Qh, KPh, VT, CTX);

  gemm_h16_kernel<<<dim3(MROWS / 128, DMODEL / 64), 256, 0, stream>>>(
      CTX, WdT, wd_b, nullptr, nullptr, out, MROWS, DMODEL, DMODEL);
}
